// NodeEncoder_66056597012589
// MI455X (gfx1250) — compile-verified, matrix operand order checked by reading, not run on hardware
//
#include <hip/hip_runtime.h>

// ---------------------------------------------------------------------------
// Types for WMMA fragments (CDNA5 wave32)
// ---------------------------------------------------------------------------
typedef __attribute__((ext_vector_type(16))) __bf16 v16bf;
typedef __attribute__((ext_vector_type(8)))  __bf16 v8bf;
typedef __attribute__((ext_vector_type(8)))  float  v8f;

#define D_FEAT 128

// ---------------------------------------------------------------------------
// GEMM: Z[npad x 128] = Hb[npad x 128](bf16) @ W (given as WT[Nout][K] bf16),
// f32 out.  8 waves / block; wave w owns output columns [w*16, w*16+16).
// B fragments (4 K-steps) stay resident in VGPRs; block walks 4 M-tiles.
// Fragment layouts per CDNA5 ISA 7.12.2 (16-bit A 16x32 / B 32x16):
//   lane l: lo = l&15 selects row (A) / col (B); hi = l>>4 selects K-halves;
//   fragment elements 0..7  -> K = 32s + hi*8 + 0..7      (contiguous)
//   fragment elements 8..15 -> K = 32s + 16 + hi*8 + 0..7 (contiguous)
// With WT[n][k] row-major, B loads are contiguous b128 pairs like A.
//   C/D (16x16 f32): lane l, vgpr v -> row = hi*8 + v, col = lo.
// ---------------------------------------------------------------------------
__global__ __launch_bounds__(256) void gcn_gemm_wmma(
    const __bf16* __restrict__ Hb, const __bf16* __restrict__ WT,
    float* __restrict__ Z) {
  const int wave = threadIdx.x >> 5;
  const int lane = threadIdx.x & 31;
  const int lo   = lane & 15;
  const int hi   = lane >> 4;
  const int n0   = wave * 16;

  // Resident B fragments for the 4 K-steps (K = 128 = 4 * 32).
  const __bf16* wrow = WT + (size_t)(n0 + lo) * D_FEAT;
  v16bf bfrag[4];
#pragma unroll
  for (int s = 0; s < 4; ++s) {
    const int k0 = s * 32;
    union { v16bf v; v8bf h[2]; } b;
    b.h[0] = *(const v8bf*)(wrow + k0 + hi * 8);
    b.h[1] = *(const v8bf*)(wrow + k0 + 16 + hi * 8);
    bfrag[s] = b.v;
  }

  const int mt0 = blockIdx.x * 4;
#pragma unroll
  for (int mt = 0; mt < 4; ++mt) {
    const int m0  = (mt0 + mt) * 16;
    const __bf16* hrow = Hb + (size_t)(m0 + lo) * D_FEAT;
    v8f c = {};
#pragma unroll
    for (int s = 0; s < 4; ++s) {
      const int k0 = s * 32;
      union { v16bf v; v8bf h[2]; } a;
      a.h[0] = *(const v8bf*)(hrow + k0 + hi * 8);
      a.h[1] = *(const v8bf*)(hrow + k0 + 16 + hi * 8);
      c = __builtin_amdgcn_wmma_f32_16x16x32_bf16(
          /*neg_a=*/false, a.v, /*neg_b=*/false, bfrag[s],
          /*c_mod=*/(short)0, c, /*reuse_a=*/false, /*reuse_b=*/false);
    }
#pragma unroll
    for (int v = 0; v < 8; ++v) {
      Z[(size_t)(m0 + hi * 8 + v) * D_FEAT + n0 + lo] = c[v];
    }
  }
}

// ---------------------------------------------------------------------------
// Degree accumulation over edges + self-loops (deg is pre-zeroed)
// ---------------------------------------------------------------------------
__global__ void degree_kernel(const int* __restrict__ ei,
                              const float* __restrict__ ew,
                              float* __restrict__ deg, int E, int N) {
  const int e = blockIdx.x * blockDim.x + threadIdx.x;
  if (e >= E + N) return;
  int   d;
  float w;
  if (e < E) { d = ei[E + e]; w = ew[e]; }
  else       { d = e - E;     w = 1.0f;  }
  atomicAdd(&deg[d], w);
}

// deg -> d^(-1/2) in place (0 where deg <= 0)
__global__ void dinv_kernel(float* __restrict__ deg, int N) {
  const int i = blockIdx.x * blockDim.x + threadIdx.x;
  if (i < N) {
    const float d = deg[i];
    deg[i] = (d > 0.0f) ? rsqrtf(d) : 0.0f;
  }
}

// ---------------------------------------------------------------------------
// Edge aggregation: one wave per edge, lane handles a float4 feature slice.
// acc[dst] += Z[src] * (dinv[src] * w * dinv[dst]).
// acc is pre-initialized with the broadcast bias row, so the epilogue is a
// pure ReLU.  All atomics land in L2 (accumulator is 25.6 MB << 192 MB L2).
// ---------------------------------------------------------------------------
__global__ __launch_bounds__(256) void scatter_kernel(
    const float* __restrict__ Z, const int* __restrict__ ei,
    const float* __restrict__ ew, const float* __restrict__ dinv,
    float* __restrict__ acc, int E, int N) {
  const int eidx = blockIdx.x * (blockDim.x >> 5) + (threadIdx.x >> 5);
  if (eidx >= E + N) return;
  const int lane = threadIdx.x & 31;

  int   s, d;
  float w;
  if (eidx < E) { s = ei[eidx]; d = ei[E + eidx]; w = ew[eidx]; }
  else          { s = eidx - E; d = s;            w = 1.0f;     }

  const float nrm = dinv[s] * w * dinv[d];
  const float4 z = *(const float4*)(Z + (size_t)s * D_FEAT + lane * 4);
  float* o = acc + (size_t)d * D_FEAT + lane * 4;
  atomicAdd(o + 0, z.x * nrm);
  atomicAdd(o + 1, z.y * nrm);
  atomicAdd(o + 2, z.z * nrm);
  atomicAdd(o + 3, z.w * nrm);
}

// ---------------------------------------------------------------------------
// Elementwise helpers
// ---------------------------------------------------------------------------
__global__ void fill_f32(float* __restrict__ p, float v, long long n) {
  const long long i = (long long)blockIdx.x * blockDim.x + threadIdx.x;
  if (i < n) p[i] = v;
}

// acc[i] = bias[i & 127]  (bias pre-seeded into accumulator)
__global__ void fill_bias(float* __restrict__ p, const float* __restrict__ bias,
                          long long n) {
  const long long i = (long long)blockIdx.x * blockDim.x + threadIdx.x;
  if (i < n) p[i] = bias[i & (D_FEAT - 1)];
}

// x(f32) -> Hb(bf16), zero-padding rows [N, npad)
__global__ void cvt_x_bf16(const float* __restrict__ x, __bf16* __restrict__ Hb,
                           long long n_valid, long long n_total) {
  const long long i = (long long)blockIdx.x * blockDim.x + threadIdx.x;
  if (i < n_total) Hb[i] = (i < n_valid) ? (__bf16)x[i] : (__bf16)0.0f;
}

// W[k][n] (f32, row-major) -> WT[n][k] (bf16) so B-fragment loads are
// contiguous 16-byte chunks in the GEMM.
__global__ void cvt_w_transpose_bf16(const float* __restrict__ W,
                                     __bf16* __restrict__ WT, int n) {
  const int i = blockIdx.x * blockDim.x + threadIdx.x;
  if (i < n) {
    const int k = i >> 7;          // row of W
    const int c = i & (D_FEAT - 1);  // col of W
    WT[(size_t)c * D_FEAT + k] = (__bf16)W[i];
  }
}

// relu -> bf16 H for next layer (bias already inside acc)
__global__ void relu_to_bf16(const float* __restrict__ acc,
                             __bf16* __restrict__ Hb, long long total) {
  const long long i = (long long)blockIdx.x * blockDim.x + threadIdx.x;
  if (i < total) Hb[i] = (__bf16)fmaxf(acc[i], 0.0f);
}

// final layer: out = relu(out), in place, f32 (bias already inside out)
__global__ void relu_f32(float* __restrict__ out, long long total) {
  const long long i = (long long)blockIdx.x * blockDim.x + threadIdx.x;
  if (i < total) out[i] = fmaxf(out[i], 0.0f);
}

// ---------------------------------------------------------------------------
// Host orchestration
// ---------------------------------------------------------------------------
static inline unsigned cdiv_u(long long a, long long b) {
  return (unsigned)((a + b - 1) / b);
}

extern "C" void kernel_launch(void* const* d_in, const int* in_sizes, int n_in,
                              void* d_out, int out_size, void* d_ws,
                              size_t ws_size, hipStream_t stream) {
  const float* x  = (const float*)d_in[0];
  const int*   ei = (const int*)d_in[1];
  const float* ew = (const float*)d_in[2];
  const float* Wm[3] = {(const float*)d_in[3], (const float*)d_in[5],
                        (const float*)d_in[7]};
  const float* bm[3] = {(const float*)d_in[4], (const float*)d_in[6],
                        (const float*)d_in[8]};
  float* out = (float*)d_out;

  const int N = in_sizes[0] / D_FEAT;
  const int E = in_sizes[2];
  const long long npad = ((long long)N + 255LL) & ~255LL;

  // Workspace carve-up
  char*   p    = (char*)d_ws;
  __bf16* Hb   = (__bf16*)p;  p += npad * D_FEAT * sizeof(__bf16);
  __bf16* Wb   = (__bf16*)p;  p += 3LL * D_FEAT * D_FEAT * sizeof(__bf16);
  float*  Z    = (float*)p;   p += npad * D_FEAT * sizeof(float);
  float*  acc  = (float*)p;   p += (long long)N * D_FEAT * sizeof(float);
  float*  dinv = (float*)p;   p += (long long)N * sizeof(float);

  const long long tot_feat = (long long)N * D_FEAT;   // N*128
  const long long tot_pad  = npad * D_FEAT;           // npad*128
  const int edges_total    = E + N;

  // --- normalization coefficients ---
  fill_f32<<<cdiv_u(N, 256), 256, 0, stream>>>(dinv, 0.0f, N);
  degree_kernel<<<cdiv_u(edges_total, 256), 256, 0, stream>>>(ei, ew, dinv, E, N);
  dinv_kernel<<<cdiv_u(N, 256), 256, 0, stream>>>(dinv, N);

  // --- precision conversion / weight repack ---
  cvt_x_bf16<<<cdiv_u(tot_pad, 256), 256, 0, stream>>>(x, Hb, tot_feat, tot_pad);
  for (int l = 0; l < 3; ++l) {
    cvt_w_transpose_bf16<<<cdiv_u(D_FEAT * D_FEAT, 256), 256, 0, stream>>>(
        Wm[l], Wb + (size_t)l * D_FEAT * D_FEAT, D_FEAT * D_FEAT);
  }

  // --- 3 GCN layers ---
  for (int l = 0; l < 3; ++l) {
    // Z = Hb @ W[l]   (WMMA bf16 -> f32)
    gcn_gemm_wmma<<<(unsigned)(npad / 64), 256, 0, stream>>>(
        Hb, Wb + (size_t)l * D_FEAT * D_FEAT, Z);

    float* accT = (l == 2) ? out : acc;
    // Seed accumulator with bias row (replaces zero-fill + later bias add)
    fill_bias<<<cdiv_u(tot_feat, 256), 256, 0, stream>>>(accT, bm[l], tot_feat);

    // accT[dst] += Z[src] * norm  (one wave per edge, L2-resident atomics)
    scatter_kernel<<<cdiv_u(edges_total, 8), 256, 0, stream>>>(
        Z, ei, ew, dinv, accT, E, N);

    if (l < 2) {
      relu_to_bf16<<<cdiv_u(tot_feat, 256), 256, 0, stream>>>(accT, Hb, tot_feat);
    } else {
      relu_f32<<<cdiv_u(tot_feat, 256), 256, 0, stream>>>(out, tot_feat);
    }
  }
}
